// PointNet2SSGCls_28432683499832
// MI455X (gfx1250) — matrix operand order repaired and checked by reading, not compiled, not run on hardware
//
#include <hip/hip_runtime.h>
#include <cstddef>

#define B_   16
#define N_   4096
#define S1_  1024
#define K1_  32
#define S2_  256
#define K2_  64
#define EPSF 1e-5f

#define HAVE_TDM (__has_builtin(__builtin_amdgcn_tensor_load_to_lds) && \
                  __has_builtin(__builtin_amdgcn_s_wait_tensorcnt))

typedef __attribute__((ext_vector_type(16))) __bf16 v16bf;
typedef __attribute__((ext_vector_type(8)))  float  v8f;
typedef unsigned int v4u __attribute__((ext_vector_type(4)));
typedef int          v4i __attribute__((ext_vector_type(4)));
typedef int          v8i __attribute__((ext_vector_type(8)));

__device__ __forceinline__ unsigned short f2bf(float f) {
  unsigned u = __float_as_uint(f);
  return (unsigned short)((u + (((u >> 16) & 1u) + 0x7FFFu)) >> 16);
}
__device__ __forceinline__ float bf2f(unsigned short h) {
  return __uint_as_float(((unsigned)h) << 16);
}

// ---------------------------------------------------------------------------
// Farthest point sampling: one block per batch; dist[] lives in LDS.
// Matches reference: start at index 0, argmax(min-dist) with first-index ties.
// ---------------------------------------------------------------------------
__global__ __launch_bounds__(256)
void fps_kernel(const float* __restrict__ xyz, float* __restrict__ new_xyz,
                int N, int S) {
  const int b = blockIdx.x;
  const int tid = threadIdx.x;
  const float* xb = xyz + (size_t)b * N * 3;
  float* ob = new_xyz + (size_t)b * S * 3;

  __shared__ float dist[N_];
  __shared__ float rv[256];
  __shared__ int   ri[256];
  __shared__ int   lastSh;

  for (int j = tid; j < N; j += 256) dist[j] = 1e10f;
  if (tid == 0) { ob[0] = xb[0]; ob[1] = xb[1]; ob[2] = xb[2]; lastSh = 0; }
  __syncthreads();

  int last = 0;
  for (int i = 1; i < S; ++i) {
    const float px = xb[last * 3 + 0];
    const float py = xb[last * 3 + 1];
    const float pz = xb[last * 3 + 2];
    float bestv = -1.0f; int bestj = 0;
    for (int j = tid; j < N; j += 256) {
      float dx = xb[j * 3 + 0] - px;
      float dy = xb[j * 3 + 1] - py;
      float dz = xb[j * 3 + 2] - pz;
      float d = dx * dx + dy * dy + dz * dz;
      float nd = dist[j];
      nd = d < nd ? d : nd;
      dist[j] = nd;
      if (nd > bestv) { bestv = nd; bestj = j; }  // ascending j -> first max kept
    }
    rv[tid] = bestv; ri[tid] = bestj;
    __syncthreads();
    for (int k = 128; k > 0; k >>= 1) {
      if (tid < k) {
        if (rv[tid + k] > rv[tid] ||
            (rv[tid + k] == rv[tid] && ri[tid + k] < ri[tid])) {
          rv[tid] = rv[tid + k]; ri[tid] = ri[tid + k];
        }
      }
      __syncthreads();
    }
    if (tid == 0) {
      int n = ri[0];
      lastSh = n;
      ob[i * 3 + 0] = xb[n * 3 + 0];
      ob[i * 3 + 1] = xb[n * 3 + 1];
      ob[i * 3 + 2] = xb[n * 3 + 2];
    }
    __syncthreads();
    last = lastSh;
  }
}

// ---------------------------------------------------------------------------
// Ball query: first K indices (ascending) within radius; pad with first hit.
// ---------------------------------------------------------------------------
__global__ __launch_bounds__(256)
void ball_query_kernel(const float* __restrict__ xyz,
                       const float* __restrict__ new_xyz,
                       int* __restrict__ idx,
                       int N, int S, int K, float r2, int total) {
  int t = blockIdx.x * 256 + threadIdx.x;
  if (t >= total) return;
  int b = t / S;
  const float* xb = xyz + (size_t)b * N * 3;
  float qx = new_xyz[(size_t)t * 3 + 0];
  float qy = new_xyz[(size_t)t * 3 + 1];
  float qz = new_xyz[(size_t)t * 3 + 2];
  int* op = idx + (size_t)t * K;
  int cnt = 0, first = -1;
  for (int j = 0; j < N && cnt < K; ++j) {
    float dx = xb[j * 3 + 0] - qx;
    float dy = xb[j * 3 + 1] - qy;
    float dz = xb[j * 3 + 2] - qz;
    float d2 = dx * dx + dy * dy + dz * dz;
    if (d2 <= r2) { if (first < 0) first = j; op[cnt++] = j; }
  }
  if (first < 0) first = 0;
  for (; cnt < K; ++cnt) op[cnt] = first;
}

// ---------------------------------------------------------------------------
// Gather + concat into zero-padded bf16 row-major A matrices.
// ---------------------------------------------------------------------------
__global__ __launch_bounds__(256)
void gather_sa1(const float* __restrict__ xyz, const float* __restrict__ new_xyz,
                const float* __restrict__ features, const int* __restrict__ idx,
                unsigned short* __restrict__ A, int total) {
  int r = blockIdx.x * 256 + threadIdx.x;
  if (r >= total) return;
  int b = r / (S1_ * K1_);
  int s = (r / K1_) % S1_;
  int j = idx[r];
  const float* p = xyz + ((size_t)b * N_ + j) * 3;
  const float* q = new_xyz + ((size_t)b * S1_ + s) * 3;
  unsigned short* row = A + (size_t)r * 32;
  row[0] = f2bf(p[0] - q[0]);
  row[1] = f2bf(p[1] - q[1]);
  row[2] = f2bf(p[2] - q[2]);
  for (int c = 0; c < 3; ++c)
    row[3 + c] = f2bf(features[((size_t)b * 3 + c) * N_ + j]);
  for (int c = 6; c < 32; ++c) row[c] = 0;
}

__global__ __launch_bounds__(256)
void gather_sa2(const float* __restrict__ xyz1, const float* __restrict__ xyz2,
                const float* __restrict__ feats1, const int* __restrict__ idx,
                unsigned short* __restrict__ A, int total) {
  int r = blockIdx.x * 256 + threadIdx.x;
  if (r >= total) return;
  int b = r / (S2_ * K2_);
  int s = (r / K2_) % S2_;
  int j = idx[r];
  const float* p = xyz1 + ((size_t)b * S1_ + j) * 3;
  const float* q = xyz2 + ((size_t)b * S2_ + s) * 3;
  unsigned short* row = A + (size_t)r * 160;
  row[0] = f2bf(p[0] - q[0]);
  row[1] = f2bf(p[1] - q[1]);
  row[2] = f2bf(p[2] - q[2]);
  const float* f = feats1 + ((size_t)b * S1_ + j) * 128;
  for (int c = 0; c < 128; ++c) row[3 + c] = f2bf(f[c]);
  for (int c = 131; c < 160; ++c) row[c] = 0;
}

__global__ __launch_bounds__(256)
void gather_sa3(const float* __restrict__ xyz2, const float* __restrict__ feats2,
                unsigned short* __restrict__ A, int total) {
  int r = blockIdx.x * 256 + threadIdx.x;
  if (r >= total) return;
  unsigned short* row = A + (size_t)r * 288;
  const float* q = xyz2 + (size_t)r * 3;
  row[0] = f2bf(q[0]); row[1] = f2bf(q[1]); row[2] = f2bf(q[2]);
  const float* f = feats2 + (size_t)r * 256;
  for (int c = 0; c < 256; ++c) row[3 + c] = f2bf(f[c]);
  for (int c = 259; c < 288; ++c) row[c] = 0;
}

// ---------------------------------------------------------------------------
// Tensor Data Mover: load a 2D bf16 tile (32 elems x 64 rows) into LDS.
// D# built per CDNA5 ISA ch. 8: group0 = count/lds_addr/global_addr/type,
// group1 = data_size, tensor dims (remaining extents -> OOB rows read zero),
// tile dims 32x64, dim0 stride. Groups 2/3 zero (2D tensor).
// ---------------------------------------------------------------------------
#if HAVE_TDM
__device__ __forceinline__ void tdm_load_tile_2d(unsigned lds_off,
                                                 const unsigned short* gptr,
                                                 int rows_rem, int cols_rem,
                                                 int stride_elems) {
  unsigned long long ga = (unsigned long long)(size_t)gptr;
  v4u g0;
  g0[0] = 1u;                                           // count=1, user mode
  g0[1] = lds_off;                                      // lds_addr (bytes)
  g0[2] = (unsigned)ga;                                 // global_addr[31:0]
  g0[3] = (unsigned)((ga >> 32) & 0x01FFFFFFu) | (2u << 30);  // [56:32] | type=2
  unsigned td0 = (unsigned)cols_rem;                    // tensor_dim0
  unsigned td1 = (unsigned)rows_rem;                    // tensor_dim1
  v8i g1;
  g1[0] = (int)(1u << 16);                              // data_size=1 -> 2 bytes
  g1[1] = (int)((td0 & 0xFFFFu) << 16);                 // tensor_dim0[15:0]
  g1[2] = (int)((td0 >> 16) | ((td1 & 0xFFFFu) << 16)); // d0[31:16] | d1[15:0]
  g1[3] = (int)((td1 >> 16) | (32u << 16));             // d1[31:16] | tile_dim0=32
  g1[4] = 64;                                           // tile_dim1=64, tile_dim2=0
  g1[5] = (int)(unsigned)stride_elems;                  // tensor_dim0_stride lo32
  g1[6] = 0;                                            // stride hi16 | d1_stride lo16
  g1[7] = 0;
  v4i z4 = {0, 0, 0, 0};
#if __clang_major__ >= 23
  v8i z8 = {0, 0, 0, 0, 0, 0, 0, 0};
  __builtin_amdgcn_tensor_load_to_lds(g0, g1, z4, z4, z8, 0);
#else
  __builtin_amdgcn_tensor_load_to_lds(g0, g1, z4, z4, 0);
#endif
}
#endif

// ---------------------------------------------------------------------------
// WMMA bf16 GEMM:  Y[M x Np] = A[M x Kp] * W[Np x Kp]^T (row-major operands).
// Block = 4 waves -> 64(M) x 64(N) output tile. Each wave: 16x64 (4 accs),
// A fragment reused across 4 WMMAs. B panel (64 rows x 32 k) staged into a
// double-buffered LDS tile by the TDM (wave 0 issues, TENSORcnt-synced),
// consumed via ds_load per the ISA 16-bit B VGPR layout.
// ---------------------------------------------------------------------------
union BFV { v16bf v; unsigned int u[8]; };
struct U4 { unsigned int x, y, z, w; };

__global__ __launch_bounds__(128)
void wmma_gemm_bf16(const unsigned short* __restrict__ A,
                    const unsigned short* __restrict__ W,
                    unsigned short* __restrict__ Y,
                    int M, int Kp, int Np) {
  __shared__ unsigned short tileB[2][64 * 32];          // 2 x 4KB
  const int lane = threadIdx.x & 31;
  const int wave = threadIdx.x >> 5;
  const int mtile = blockIdx.x * 4 + wave;
  const bool active = (mtile * 16) < M;                 // wave-uniform
  const int mbase = active ? mtile * 16 : 0;
  const int nbase = blockIdx.y * 64;
  const int half = lane >> 4;
  const int l15 = lane & 15;

  const unsigned short* arow = A + (size_t)(mbase + l15) * Kp;
  v8f acc[4] = {v8f{}, v8f{}, v8f{}, v8f{}};

  const int nsteps = Kp >> 5;
#if HAVE_TDM
  if (wave == 0)
    tdm_load_tile_2d((unsigned)(size_t)&tileB[0][0], W + (size_t)nbase * Kp,
                     Np - nbase, Kp, Kp);
#endif
  for (int s = 0; s < nsteps; ++s) {
    const int k0 = s << 5;
    const int cur = s & 1;
#if HAVE_TDM
    if (wave == 0) {
      if (s + 1 < nsteps) {
        tdm_load_tile_2d((unsigned)(size_t)&tileB[cur ^ 1][0],
                         W + (size_t)nbase * Kp + (k0 + 32),
                         Np - nbase, Kp - (k0 + 32), Kp);
        __builtin_amdgcn_s_wait_tensorcnt(1);   // older (current buf) done
      } else {
        __builtin_amdgcn_s_wait_tensorcnt(0);   // drain on last step
      }
    }
    __syncthreads();
#else
    {  // cooperative staging fallback: 128 threads x 32B
      int r = threadIdx.x >> 1;
      int h = (threadIdx.x & 1) << 4;
      unsigned short* dst = &tileB[cur][r * 32 + h];
      int grow = nbase + r;
      if (grow < Np) {
        const unsigned short* src = W + (size_t)grow * Kp + k0 + h;
        *(U4*)dst = *(const U4*)src;
        *(U4*)(dst + 8) = *(const U4*)(src + 8);
      } else {
        U4 z = {0, 0, 0, 0};
        *(U4*)dst = z; *(U4*)(dst + 8) = z;
      }
    }
    __syncthreads();
#endif
    if (active) {
      if (s + 1 < nsteps) __builtin_prefetch(arow + k0 + 32, 0, 0);
      BFV a;
#pragma unroll
      for (int j = 0; j < 8; ++j) {
        // A 16x32 bf16 layout: VGPR j holds K pair base + 2*(j&3),
        // base = k0 + ((j>=4)?16:0) + half*8
        int ka = k0 + ((j & 4) << 2) + half * 8 + ((j & 3) << 1);
        a.u[j] = *(const unsigned int*)(arow + ka);
      }
#pragma unroll
      for (int t = 0; t < 4; ++t) {
        BFV b;
        // B 32x16 bf16 layout: lane -> col n, K = half*8 + {0..7} and 16 + half*8 + {0..7}.
        const unsigned short* bp = &tileB[cur][(t * 16 + l15) * 32 + half * 8];
        *(U4*)&b.u[0] = *(const U4*)bp;
        *(U4*)&b.u[4] = *(const U4*)(bp + 16);
        acc[t] = __builtin_amdgcn_wmma_f32_16x16x32_bf16(
            false, a.v, false, b.v, (short)0, acc[t], false, false);
      }
    }
    __syncthreads();
  }
  if (active) {
#pragma unroll
    for (int t = 0; t < 4; ++t) {
      if (nbase + t * 16 < Np) {            // wave-uniform subtile guard
        unsigned short* yp = Y + nbase + t * 16 + l15;
#pragma unroll
        for (int j = 0; j < 8; ++j) {
          int m = mbase + j + 8 * half;     // D: VGPR j -> row j + 8*half
          yp[(size_t)m * Np] = f2bf(acc[t][j]);
        }
      }
    }
  }
}

// ---------------------------------------------------------------------------
// BatchNorm over all M rows (axes (0,1,2) of (B,S,K,C)) + ReLU, bf16 in/out.
// ---------------------------------------------------------------------------
__global__ __launch_bounds__(256)
void zero_kernel(float* __restrict__ p, int n) {
  int i = blockIdx.x * 256 + threadIdx.x;
  if (i < n) p[i] = 0.0f;
}

__global__ __launch_bounds__(256)
void bn_reduce(const unsigned short* __restrict__ Y, float* __restrict__ sums,
               int M, int N) {
  const int n = blockIdx.x;
  const int tid = threadIdx.x;
  float s = 0.0f, s2 = 0.0f;
  for (int r = blockIdx.y * 256 + tid; r < M; r += gridDim.y * 256) {
    float v = bf2f(Y[(size_t)r * N + n]);
    s += v; s2 += v * v;
  }
  __shared__ float ls[256], lq[256];
  ls[tid] = s; lq[tid] = s2;
  __syncthreads();
  for (int k = 128; k > 0; k >>= 1) {
    if (tid < k) { ls[tid] += ls[tid + k]; lq[tid] += lq[tid + k]; }
    __syncthreads();
  }
  if (tid == 0) {
    atomicAdd(&sums[n], ls[0]);
    atomicAdd(&sums[N + n], lq[0]);
  }
}

__global__ __launch_bounds__(256)
void bn_apply_relu(const unsigned short* __restrict__ Yin,
                   unsigned short* __restrict__ Yout,
                   const float* __restrict__ sums,
                   const float* __restrict__ gamma,
                   const float* __restrict__ beta,
                   size_t total, int N, float invM) {
  size_t i = (size_t)blockIdx.x * 256 + threadIdx.x;
  if (i >= total) return;
  int n = (int)(i % (size_t)N);
  float mean = sums[n] * invM;
  float var = sums[N + n] * invM - mean * mean;
  float v = bf2f(Yin[i]);
  v = gamma[n] * (v - mean) * rsqrtf(var + EPSF) + beta[n];
  v = v > 0.0f ? v : 0.0f;
  Yout[i] = f2bf(v);
}

// Max over sample dimension: in = G groups of K rows x N channels (bf16).
__global__ __launch_bounds__(256)
void maxpool_kernel(const unsigned short* __restrict__ Yin,
                    float* __restrict__ out, int G, int K, int N) {
  int i = blockIdx.x * 256 + threadIdx.x;
  if (i >= G * N) return;
  int g = i / N, n = i % N;
  const unsigned short* p = Yin + (size_t)g * K * N + n;
  float m = -1e30f;
  for (int k = 0; k < K; ++k) {
    float v = bf2f(p[(size_t)k * N]);
    m = v > m ? v : m;
  }
  out[i] = m;
}

__global__ __launch_bounds__(256)
void cast_bf16(const float* __restrict__ in, unsigned short* __restrict__ out,
               int n) {
  int i = blockIdx.x * 256 + threadIdx.x;
  if (i < n) out[i] = f2bf(in[i]);
}

// Convert/zero-pad fp32 weight (Cout x Cin) -> bf16 (Np x Kp).
__global__ __launch_bounds__(256)
void convw(const float* __restrict__ W, unsigned short* __restrict__ Wb,
           int Cout, int Cin, int Np, int Kp) {
  int i = blockIdx.x * 256 + threadIdx.x;
  if (i >= Np * Kp) return;
  int n = i / Kp, k = i % Kp;
  Wb[i] = (n < Cout && k < Cin) ? f2bf(W[(size_t)n * Cin + k]) : (unsigned short)0;
}

__global__ __launch_bounds__(256)
void final_out(const unsigned short* __restrict__ Y,
               const float* __restrict__ bias, float* __restrict__ out) {
  int i = blockIdx.x * 256 + threadIdx.x;
  if (i >= B_ * 40) return;
  int b = i / 40, c = i % 40;
  out[i] = bf2f(Y[b * 48 + c]) + bias[c];
}

// ---------------------------------------------------------------------------
// Host-side orchestration.
// ---------------------------------------------------------------------------
static inline int cdiv(int a, int b) { return (a + b - 1) / b; }

extern "C" void kernel_launch(void* const* d_in, const int* in_sizes, int n_in,
                              void* d_out, int out_size, void* d_ws, size_t ws_size,
                              hipStream_t stream) {
  (void)in_sizes; (void)n_in; (void)out_size; (void)ws_size;
  // Input order: xyz, mask, features, then params flattened in JAX sorted-dict
  // pytree order.
  const float* xyz      = (const float*)d_in[0];
  const float* features = (const float*)d_in[2];
  const float* bn1_b = (const float*)d_in[3];
  const float* bn1_g = (const float*)d_in[4];
  const float* bn2_b = (const float*)d_in[5];
  const float* bn2_g = (const float*)d_in[6];
  const float* fc1_w = (const float*)d_in[7];
  const float* fc2_w = (const float*)d_in[8];
  const float* fc3_b = (const float*)d_in[9];
  const float* fc3_w = (const float*)d_in[10];
  // sa layers: each layer dict flattens as b, g, w
  const float* sa_b[9], *sa_g[9], *sa_w[9];
  for (int i = 0; i < 9; ++i) {
    sa_b[i] = (const float*)d_in[11 + 3 * i + 0];
    sa_g[i] = (const float*)d_in[11 + 3 * i + 1];
    sa_w[i] = (const float*)d_in[11 + 3 * i + 2];
  }
  float* out = (float*)d_out;

  // Workspace carve-up (deterministic every call).
  size_t off = 0;
  char* base = (char*)d_ws;
  auto alloc = [&](size_t bytes) -> char* {
    char* p = base + off;
    off += (bytes + 255) & ~(size_t)255;
    return p;
  };
  const int M1 = B_ * S1_ * K1_;   // 524288
  const int M2 = B_ * S2_ * K2_;   // 262144
  const int M3 = B_ * S2_;         // 4096
  unsigned short* bufP = (unsigned short*)alloc((size_t)M1 * 128 * 2);
  unsigned short* bufQ = (unsigned short*)alloc((size_t)M1 * 128 * 2);
  int*   idx1  = (int*)  alloc((size_t)M1 * 4);
  int*   idx2  = (int*)  alloc((size_t)M2 * 4);
  float* xyz1  = (float*)alloc((size_t)B_ * S1_ * 3 * 4);
  float* xyz2  = (float*)alloc((size_t)B_ * S2_ * 3 * 4);
  float* feats1= (float*)alloc((size_t)B_ * S1_ * 128 * 4);
  float* feats2= (float*)alloc((size_t)B_ * S2_ * 256 * 4);
  float* xfc   = (float*)alloc((size_t)B_ * 1024 * 4);
  float* sums  = (float*)alloc((size_t)2 * 1024 * 4);
  // bf16 weights (Np x Kp, padded)
  struct WSpec { const float* w; int co, ci, np, kp; unsigned short* dst; };
  WSpec ws[12] = {
    {sa_w[0],  64,   6,  64,  32, nullptr}, {sa_w[1],  64,  64,  64,  64, nullptr},
    {sa_w[2], 128,  64, 128,  64, nullptr}, {sa_w[3], 128, 131, 128, 160, nullptr},
    {sa_w[4], 128, 128, 128, 128, nullptr}, {sa_w[5], 256, 128, 256, 128, nullptr},
    {sa_w[6], 256, 259, 256, 288, nullptr}, {sa_w[7], 512, 256, 512, 256, nullptr},
    {sa_w[8],1024, 512,1024, 512, nullptr}, {fc1_w,  512,1024, 512,1024, nullptr},
    {fc2_w,  256, 512, 256, 512, nullptr},  {fc3_w,   40, 256,  48, 256, nullptr},
  };
  for (int i = 0; i < 12; ++i)
    ws[i].dst = (unsigned short*)alloc((size_t)ws[i].np * ws[i].kp * 2);

  // Convert all weights to padded bf16.
  for (int i = 0; i < 12; ++i) {
    int tot = ws[i].np * ws[i].kp;
    convw<<<cdiv(tot, 256), 256, 0, stream>>>(ws[i].w, ws[i].dst,
                                              ws[i].co, ws[i].ci, ws[i].np, ws[i].kp);
  }

  auto gemm = [&](const unsigned short* A, const unsigned short* W,
                  unsigned short* Y, int M, int Kp, int Np) {
    dim3 grid(cdiv(M / 16, 4), cdiv(Np, 64));
    wmma_gemm_bf16<<<grid, 128, 0, stream>>>(A, W, Y, M, Kp, Np);
  };
  auto bn = [&](const unsigned short* Yin, unsigned short* Yout,
                const float* g, const float* b, int M, int N) {
    zero_kernel<<<cdiv(2 * N, 256), 256, 0, stream>>>(sums, 2 * N);
    int gy = cdiv(M, 256); if (gy > 64) gy = 64;
    bn_reduce<<<dim3(N, gy), 256, 0, stream>>>(Yin, sums, M, N);
    size_t tot = (size_t)M * N;
    bn_apply_relu<<<(unsigned)((tot + 255) / 256), 256, 0, stream>>>(
        Yin, Yout, sums, g, b, tot, N, 1.0f / (float)M);
  };

  // ---- SA1 ----
  fps_kernel<<<B_, 256, 0, stream>>>(xyz, xyz1, N_, S1_);
  ball_query_kernel<<<cdiv(B_ * S1_, 256), 256, 0, stream>>>(
      xyz, xyz1, idx1, N_, S1_, K1_, 0.2f * 0.2f, B_ * S1_);
  gather_sa1<<<cdiv(M1, 256), 256, 0, stream>>>(xyz, xyz1, features, idx1, bufP, M1);
  gemm(bufP, ws[0].dst, bufQ, M1, 32, 64);   bn(bufQ, bufP, sa_g[0], sa_b[0], M1, 64);
  gemm(bufP, ws[1].dst, bufQ, M1, 64, 64);   bn(bufQ, bufP, sa_g[1], sa_b[1], M1, 64);
  gemm(bufP, ws[2].dst, bufQ, M1, 64, 128);  bn(bufQ, bufP, sa_g[2], sa_b[2], M1, 128);
  maxpool_kernel<<<cdiv(B_ * S1_ * 128, 256), 256, 0, stream>>>(
      bufP, feats1, B_ * S1_, K1_, 128);

  // ---- SA2 ----
  fps_kernel<<<B_, 256, 0, stream>>>(xyz1, xyz2, S1_, S2_);
  ball_query_kernel<<<cdiv(B_ * S2_, 256), 256, 0, stream>>>(
      xyz1, xyz2, idx2, S1_, S2_, K2_, 0.4f * 0.4f, B_ * S2_);
  gather_sa2<<<cdiv(M2, 256), 256, 0, stream>>>(xyz1, xyz2, feats1, idx2, bufP, M2);
  gemm(bufP, ws[3].dst, bufQ, M2, 160, 128); bn(bufQ, bufP, sa_g[3], sa_b[3], M2, 128);
  gemm(bufP, ws[4].dst, bufQ, M2, 128, 128); bn(bufQ, bufP, sa_g[4], sa_b[4], M2, 128);
  gemm(bufP, ws[5].dst, bufQ, M2, 128, 256); bn(bufQ, bufP, sa_g[5], sa_b[5], M2, 256);
  maxpool_kernel<<<cdiv(B_ * S2_ * 256, 256), 256, 0, stream>>>(
      bufP, feats2, B_ * S2_, K2_, 256);

  // ---- SA3 (group-all) ----
  gather_sa3<<<cdiv(M3, 256), 256, 0, stream>>>(xyz2, feats2, bufP, M3);
  gemm(bufP, ws[6].dst, bufQ, M3, 288, 256); bn(bufQ, bufP, sa_g[6], sa_b[6], M3, 256);
  gemm(bufP, ws[7].dst, bufQ, M3, 256, 512); bn(bufQ, bufP, sa_g[7], sa_b[7], M3, 512);
  gemm(bufP, ws[8].dst, bufQ, M3, 512, 1024);bn(bufQ, bufP, sa_g[8], sa_b[8], M3, 1024);
  maxpool_kernel<<<cdiv(B_ * 1024, 256), 256, 0, stream>>>(
      bufP, xfc, B_, S2_, 1024);

  // ---- FC head ----
  cast_bf16<<<cdiv(B_ * 1024, 256), 256, 0, stream>>>(xfc, bufP, B_ * 1024);
  gemm(bufP, ws[9].dst, bufQ, B_, 1024, 512); bn(bufQ, bufP, bn1_g, bn1_b, B_, 512);
  gemm(bufP, ws[10].dst, bufQ, B_, 512, 256); bn(bufQ, bufP, bn2_g, bn2_b, B_, 256);
  gemm(bufP, ws[11].dst, bufQ, B_, 256, 48);
  final_out<<<cdiv(B_ * 40, 256), 256, 0, stream>>>(bufQ, fc3_b, out);
}
